// ModuleAttention_61366492725501
// MI455X (gfx1250) — matrix operand order repaired and checked by reading, not compiled, not run on hardware
//
#include <hip/hip_runtime.h>

typedef __bf16 bf16_t;
typedef __attribute__((ext_vector_type(16))) __bf16 v16bf;
typedef __attribute__((ext_vector_type(8)))  __bf16 v8bf;
typedef __attribute__((ext_vector_type(8)))  float  v8f;

static constexpr int kB    = 2;
static constexpr int kT    = 2048;
static constexpr int kH    = 1536;
static constexpr int kNH   = 8;
static constexpr int kHD   = 256;
static constexpr int kHalf = 128;

#define WMMA_BF16(a, b, c) \
  __builtin_amdgcn_wmma_f32_16x16x32_bf16(false, (a), false, (b), (short)0, (c), false, false)

__device__ __forceinline__ v8f zero_v8f() {
  v8f z = {0.f, 0.f, 0.f, 0.f, 0.f, 0.f, 0.f, 0.f};
  return z;
}

// ---- CDNA5 async global->LDS copy (ASYNCcnt-tracked) ----------------------
// the LDS destination address comes from a VGPR; generic pointers to __shared__ carry the
// LDS byte offset in their low 32 bits.
__device__ __forceinline__ void async_copy_b128(unsigned lds_addr, const void* gaddr) {
  asm volatile("global_load_async_to_lds_b128 %0, %1, off"
               :: "v"(lds_addr), "v"(gaddr) : "memory");
}
__device__ __forceinline__ void wait_asynccnt0() {
  asm volatile("s_wait_asynccnt 0x0" ::: "memory");
}

// A-fragment (16x32 bf16): lane = row (lane&15), half h = lane>>4.
// elems j=0..7  -> K = h*16 + j       (contiguous 16B)
// elems j=8..15 -> K = h*16 + j       (contiguous 16B)
__device__ __forceinline__ v16bf wmma_a_frag(const bf16_t* base, int ld) {
  const int lane = threadIdx.x & 31;
  const int m = lane & 15;
  const int h = lane >> 4;
  const bf16_t* row = base + m * ld;
  v8bf lo = *(const v8bf*)(row + h * 16);
  v8bf hi = *(const v8bf*)(row + h * 16 + 8);
  return __builtin_shufflevector(lo, hi, 0, 1, 2, 3, 4, 5, 6, 7,
                                 8, 9, 10, 11, 12, 13, 14, 15);
}

// B-fragment (32x16 bf16): lane = column (lane&15), half h = lane>>4.
// elems j=0..15 -> K = h*16 + j (contiguous 32B). "base" points at a
// row-major [N][K] tile (i.e. the B matrix stored column-major).
__device__ __forceinline__ v16bf wmma_b_frag(const bf16_t* base, int ld) {
  const int lane = threadIdx.x & 31;
  const int n = lane & 15;
  const int h = lane >> 4;
  return *(const v16bf*)(base + n * ld + h * 16);
}

// ---------------------------------------------------------------------------
// GEMM: Y(MxN, f32) = X(MxK, f32) @ W(KxN, f32); bf16 WMMA compute.
// Block: 256 threads (8 waves). Tile: 128(M) x 128(N), K-step 32.
// Wave w computes rows [w*16, w*16+16) x all 128 N columns (8 C fragments).
// All 8 B-fragments are loaded in one clause, then 8 independent WMMAs issue
// back-to-back behind a single s_wait_dscnt.
// ---------------------------------------------------------------------------
__global__ __launch_bounds__(256) void gemm_bf16_kernel(
    const float* __restrict__ X, const float* __restrict__ W,
    float* __restrict__ Y, int M, int N, int K) {
  __shared__ __attribute__((aligned(32))) bf16_t Xs[128][32];
  __shared__ __attribute__((aligned(32))) bf16_t Wts[128][32];  // transposed W tile

  const int tid  = threadIdx.x;
  const int wave = tid >> 5;
  const int lane = tid & 31;
  const int n0 = blockIdx.x * 128;
  const int m0 = blockIdx.y * 128;

  // X-tile mapping: row xr = tid/2, 16-col half xc = (tid&1)*16
  const int xr = tid >> 1, xc = (tid & 1) * 16;
  // W-tile mapping: column wn = tid&127, 16-row k half wk = (tid>>7)*16
  const int wn = tid & 127, wk = (tid >> 7) * 16;

  v8f acc[8];
#pragma unroll
  for (int st = 0; st < 8; ++st) acc[st] = zero_v8f();

  for (int k0 = 0; k0 < K; k0 += 32) {
    const float* xsrc = X + (long)(m0 + xr) * K + (k0 + xc);
    const float* wsrc = W + (long)(k0 + wk) * N + (n0 + wn);
    if (k0 + 32 < K) {  // prefetch next tiles (global_prefetch_b8)
      __builtin_prefetch(xsrc + 32, 0, 1);
      __builtin_prefetch(wsrc + 32 * (long)N, 0, 1);
    }
    // X tile 128x32: 16 contiguous f32 per thread -> two packed 16B stores
#pragma unroll
    for (int half = 0; half < 2; ++half) {
      const float4* xs4 = (const float4*)xsrc;
      float4 f0 = xs4[half * 2 + 0];
      float4 f1 = xs4[half * 2 + 1];
      v8bf pk;
      pk[0] = (bf16_t)f0.x; pk[1] = (bf16_t)f0.y;
      pk[2] = (bf16_t)f0.z; pk[3] = (bf16_t)f0.w;
      pk[4] = (bf16_t)f1.x; pk[5] = (bf16_t)f1.y;
      pk[6] = (bf16_t)f1.z; pk[7] = (bf16_t)f1.w;
      *(v8bf*)&Xs[xr][xc + half * 8] = pk;
    }
    // W tile 32x128 -> transposed: thread gathers 16 k-values at fixed n
#pragma unroll
    for (int half = 0; half < 2; ++half) {
      v8bf pk;
#pragma unroll
      for (int i = 0; i < 8; ++i)
        pk[i] = (bf16_t)wsrc[(long)(half * 8 + i) * N];
      *(v8bf*)&Wts[wn][wk + half * 8] = pk;
    }
    __syncthreads();

    // Batch fragment loads, then a burst of 8 independent WMMAs.
    v16bf a = wmma_a_frag(&Xs[wave * 16][0], 32);
    v16bf bfr[8];
#pragma unroll
    for (int st = 0; st < 8; ++st) bfr[st] = wmma_b_frag(&Wts[st * 16][0], 32);
#pragma unroll
    for (int st = 0; st < 8; ++st) acc[st] = WMMA_BF16(a, bfr[st], acc[st]);
    __syncthreads();
  }

  // C layout: VGPR r -> M = r + 8*(lane>>4); N = lane&15
  const int n = lane & 15;
  const int h = lane >> 4;
#pragma unroll
  for (int st = 0; st < 8; ++st) {
#pragma unroll
    for (int r = 0; r < 8; ++r) {
      int m = r + 8 * h;
      Y[(long)(m0 + wave * 16 + m) * N + (n0 + st * 16 + n)] = acc[st][r];
    }
  }
}

// ---------------------------------------------------------------------------
// RMSNorm + RoPE + bf16 pack.
// Grid: B*T blocks, 288 threads (9 waves). Waves 0..7: q heads. Wave 8: k + v.
// q is additionally scaled by 1/sqrt(HD) so attention skips the scale.
// Outputs: qh (B,NH,T,HD) bf16; kh,vh (B,T,HD) bf16.
// ---------------------------------------------------------------------------
__global__ __launch_bounds__(288) void normrope_kernel(
    const float* __restrict__ qb, const float* __restrict__ kb,
    const float* __restrict__ vb, const int* __restrict__ positions,
    const float* __restrict__ cosT, const float* __restrict__ sinT,
    const float* __restrict__ qw, const float* __restrict__ kw,
    bf16_t* __restrict__ qh, bf16_t* __restrict__ kh, bf16_t* __restrict__ vh) {
  const int bt   = blockIdx.x;
  const int b    = bt / kT;
  const int t    = bt % kT;
  const int wave = threadIdx.x >> 5;
  const int lane = threadIdx.x & 31;
  const int pos  = positions[t];
  const float attn_scale = 0.0625f;  // 1/sqrt(256)

  if (wave < kNH) {
    const float* src = qb + (long)bt * (kNH * kHD) + wave * kHD;
    float x[8];
    float ss = 0.f;
#pragma unroll
    for (int e = 0; e < 8; ++e) {
      x[e] = src[lane * 8 + e];
      ss += x[e] * x[e];
    }
#pragma unroll
    for (int off = 16; off >= 1; off >>= 1) ss += __shfl_xor(ss, off, 32);
    float rn = rsqrtf(ss * (1.0f / kHD) + 1e-6f);

    v8bf ov;
#pragma unroll
    for (int e = 0; e < 8; ++e) {
      int d  = lane * 8 + e;
      float xn = x[e] * rn * qw[d];
      float pr = __shfl_xor(xn, 16, 32);  // partner: d ^ 128
      int dh = d & (kHalf - 1);
      float c = cosT[(long)pos * kHalf + dh];
      float s = sinT[(long)pos * kHalf + dh];
      float o = (d < kHalf) ? (xn * c - pr * s) : (xn * c + pr * s);
      ov[e] = (bf16_t)(o * attn_scale);
    }
    *(v8bf*)(qh + (((long)b * kNH + wave) * kT + t) * kHD + lane * 8) = ov;
  } else {
    // k: rmsnorm + rope
    const float* src = kb + (long)bt * kHD;
    float x[8];
    float ss = 0.f;
#pragma unroll
    for (int e = 0; e < 8; ++e) {
      x[e] = src[lane * 8 + e];
      ss += x[e] * x[e];
    }
#pragma unroll
    for (int off = 16; off >= 1; off >>= 1) ss += __shfl_xor(ss, off, 32);
    float rn = rsqrtf(ss * (1.0f / kHD) + 1e-6f);

    v8bf ov;
#pragma unroll
    for (int e = 0; e < 8; ++e) {
      int d  = lane * 8 + e;
      float xn = x[e] * rn * kw[d];
      float pr = __shfl_xor(xn, 16, 32);
      int dh = d & (kHalf - 1);
      float c = cosT[(long)pos * kHalf + dh];
      float s = sinT[(long)pos * kHalf + dh];
      float o = (d < kHalf) ? (xn * c - pr * s) : (xn * c + pr * s);
      ov[e] = (bf16_t)o;
    }
    *(v8bf*)(kh + (long)bt * kHD + lane * 8) = ov;
    // v: plain f32 -> bf16, packed store
    const float* vs = vb + (long)bt * kHD;
    v8bf vv;
#pragma unroll
    for (int e = 0; e < 8; ++e) vv[e] = (bf16_t)vs[lane * 8 + e];
    *(v8bf*)(vh + (long)bt * kHD + lane * 8) = vv;
  }
}

// ---------------------------------------------------------------------------
// Flash attention (causal, GQA NKV=1 so all heads share K/V).
// Block: 128 threads (4 waves). Each wave: 16 query rows; block: 64 queries.
// Q/K tiles staged via async global->LDS (ASYNCcnt); V transposed manually.
// Online softmax; O accum in 16 v8f fragments (f32).
// Output: attn (B,T,NH*HD) f32 for the Wo GEMM.
// ---------------------------------------------------------------------------
__global__ __launch_bounds__(128) void attn_kernel(
    const bf16_t* __restrict__ qh, const bf16_t* __restrict__ kh,
    const bf16_t* __restrict__ vh, float* __restrict__ out) {
  __shared__ __attribute__((aligned(32))) bf16_t Qs[64][kHD];   // 32 KB
  __shared__ __attribute__((aligned(32))) bf16_t Ks[64][kHD];   // 32 KB
  __shared__ __attribute__((aligned(32))) bf16_t Vst[kHD][64];  // 32 KB (transposed)
  __shared__ __attribute__((aligned(32))) bf16_t Ps[4][16][64]; // 8 KB

  const int tid  = threadIdx.x;
  const int wave = tid >> 5;
  const int lane = tid & 31;
  const int nt   = lane & 15;
  const int h    = lane >> 4;

  const int ntiles = kT / 64;
  const int qtile  = blockIdx.x % ntiles;
  const int hb     = blockIdx.x / ntiles;
  const int head   = hb % kNH;
  const int b      = hb / kNH;
  const int q0     = qtile * 64;

  const bf16_t* qptr = qh + (((long)b * kNH + head) * kT + q0) * kHD;
  const bf16_t* kptr = kh + (long)b * kT * kHD;
  const bf16_t* vptr = vh + (long)b * kT * kHD;

  const unsigned qs_lds = (unsigned)(uintptr_t)&Qs[0][0];
  const unsigned ks_lds = (unsigned)(uintptr_t)&Ks[0][0];

  // Async-load Q tile (64x256 bf16 = 2048 x 16B transfers).
  for (int i = tid; i < 64 * kHD / 8; i += 128)
    async_copy_b128(qs_lds + i * 16u, (const char*)qptr + (size_t)i * 16);
  wait_asynccnt0();

  v8f o[16];
#pragma unroll
  for (int i = 0; i < 16; ++i) o[i] = zero_v8f();
  float mrow[8], lrow[8];
#pragma unroll
  for (int r = 0; r < 8; ++r) { mrow[r] = -3.0e38f; lrow[r] = 0.f; }

  const int nkb = q0 / 64 + 1;  // causal: cover keys up to q0+63
  for (int kb = 0; kb < nkb; ++kb) {
    const int k0 = kb * 64;
    __syncthreads();  // Ks/Vst safe to overwrite; also fences Q-tile visibility
    // Async-load K tile.
    {
      const char* src = (const char*)(kptr + (long)k0 * kHD);
      for (int i = tid; i < 64 * kHD / 8; i += 128)
        async_copy_b128(ks_lds + i * 16u, src + (size_t)i * 16);
    }
    // V tile, transposed into Vst[d][key].
    {
      const bf16_t* src = vptr + (long)k0 * kHD;
      for (int i = tid; i < 64 * kHD; i += 128) {
        int key = i >> 8, d = i & 255;
        Vst[d][key] = src[i];
      }
    }
    wait_asynccnt0();
    __syncthreads();

    // S = Q @ K^T : 16 queries x 64 keys, reduce over HD=256.
    // Per kc: batch A + 4 B loads, then 4 independent WMMAs.
    v8f sfrag[4];
#pragma unroll
    for (int st = 0; st < 4; ++st) sfrag[st] = zero_v8f();
#pragma unroll
    for (int kc = 0; kc < 8; ++kc) {
      v16bf a = wmma_a_frag(&Qs[wave * 16][kc * 32], kHD);
      v16bf bb[4];
#pragma unroll
      for (int st = 0; st < 4; ++st)
        bb[st] = wmma_b_frag(&Ks[st * 16][kc * 32], kHD);
#pragma unroll
      for (int st = 0; st < 4; ++st)
        sfrag[st] = WMMA_BF16(a, bb[st], sfrag[st]);
    }

    // Causal mask + running row max (rows span the 16 lanes of one half-wave).
    float mnew[8];
#pragma unroll
    for (int r = 0; r < 8; ++r) mnew[r] = mrow[r];
#pragma unroll
    for (int st = 0; st < 4; ++st) {
      int key = k0 + st * 16 + nt;
#pragma unroll
      for (int r = 0; r < 8; ++r) {
        int q = q0 + wave * 16 + r + 8 * h;
        float s = sfrag[st][r];
        if (key > q) s = -1.0e9f;
        sfrag[st][r] = s;
        mnew[r] = fmaxf(mnew[r], s);
      }
    }
#pragma unroll
    for (int off = 8; off >= 1; off >>= 1) {
#pragma unroll
      for (int r = 0; r < 8; ++r)
        mnew[r] = fmaxf(mnew[r], __shfl_xor(mnew[r], off, 32));
    }

    float cscale[8], rsum[8];
#pragma unroll
    for (int r = 0; r < 8; ++r) {
      cscale[r] = __expf(mrow[r] - mnew[r]);
      mrow[r]   = mnew[r];
      rsum[r]   = 0.f;
    }
#pragma unroll
    for (int st = 0; st < 4; ++st) {
#pragma unroll
      for (int r = 0; r < 8; ++r) {
        float p = __expf(sfrag[st][r] - mnew[r]);
        sfrag[st][r] = p;
        rsum[r] += p;
      }
    }
#pragma unroll
    for (int off = 8; off >= 1; off >>= 1) {
#pragma unroll
      for (int r = 0; r < 8; ++r) rsum[r] += __shfl_xor(rsum[r], off, 32);
    }
#pragma unroll
    for (int r = 0; r < 8; ++r) lrow[r] = lrow[r] * cscale[r] + rsum[r];

    // Rescale O accumulators.
#pragma unroll
    for (int i = 0; i < 16; ++i) {
#pragma unroll
      for (int r = 0; r < 8; ++r) o[i][r] *= cscale[r];
    }

    // P (C-layout) -> LDS -> A-layout for PV. Wave-private region; LDS ops
    // from one wave are in-order.
#pragma unroll
    for (int st = 0; st < 4; ++st) {
#pragma unroll
      for (int r = 0; r < 8; ++r)
        Ps[wave][r + 8 * h][st * 16 + nt] = (bf16_t)sfrag[st][r];
    }

    // O += P @ V
    v16bf pa0 = wmma_a_frag(&Ps[wave][0][0], 64);
    v16bf pa1 = wmma_a_frag(&Ps[wave][0][32], 64);
#pragma unroll
    for (int hdt = 0; hdt < 16; ++hdt) {
      v16bf b0 = wmma_b_frag(&Vst[hdt * 16][0], 64);
      v16bf b1 = wmma_b_frag(&Vst[hdt * 16][32], 64);
      o[hdt] = WMMA_BF16(pa0, b0, o[hdt]);
      o[hdt] = WMMA_BF16(pa1, b1, o[hdt]);
    }
  }

  // Epilogue: O / l, store f32 into (B,T,NH*HD).
#pragma unroll
  for (int r = 0; r < 8; ++r) lrow[r] = 1.0f / lrow[r];
#pragma unroll
  for (int hdt = 0; hdt < 16; ++hdt) {
#pragma unroll
    for (int r = 0; r < 8; ++r) {
      int q = q0 + wave * 16 + r + 8 * h;
      int d = hdt * 16 + nt;
      out[(long)(b * kT + q) * (kNH * kHD) + head * kHD + d] = o[hdt][r] * lrow[r];
    }
  }
}

// ---------------------------------------------------------------------------
extern "C" void kernel_launch(void* const* d_in, const int* in_sizes, int n_in,
                              void* d_out, int out_size, void* d_ws, size_t ws_size,
                              hipStream_t stream) {
  (void)in_sizes; (void)n_in; (void)out_size; (void)ws_size;

  const float* x         = (const float*)d_in[0];
  const int*   positions = (const int*)d_in[1];
  const float* cosT      = (const float*)d_in[2];
  const float* sinT      = (const float*)d_in[3];
  // d_in[4] = causal_mask (unused; causality applied analytically)
  const float* Wq = (const float*)d_in[5];
  const float* Wk = (const float*)d_in[6];
  const float* Wv = (const float*)d_in[7];
  const float* Wo = (const float*)d_in[8];
  const float* qw = (const float*)d_in[9];
  const float* kw = (const float*)d_in[10];
  float* out = (float*)d_out;

  const int M  = kB * kT;        // 4096
  const int Nq = kNH * kHD;      // 2048

  char* ws = (char*)d_ws;
  size_t off = 0;
  auto alloc = [&](size_t bytes) -> void* {
    void* p = (void*)(ws + off);
    off += (bytes + 255) & ~(size_t)255;
    return p;
  };
  float*  q_buf = (float*)alloc((size_t)M * Nq * sizeof(float));     // 32 MB
  float*  k_buf = (float*)alloc((size_t)M * kHD * sizeof(float));    // 4 MB
  float*  v_buf = (float*)alloc((size_t)M * kHD * sizeof(float));    // 4 MB
  bf16_t* qh    = (bf16_t*)alloc((size_t)M * Nq * sizeof(bf16_t));   // 16 MB
  bf16_t* kh    = (bf16_t*)alloc((size_t)M * kHD * sizeof(bf16_t));  // 2 MB
  bf16_t* vh    = (bf16_t*)alloc((size_t)M * kHD * sizeof(bf16_t));  // 2 MB
  float* attn_buf = q_buf;  // q_buf is dead after normrope; reuse for attn out

  dim3 blk256(256);
  // QKV projections
  gemm_bf16_kernel<<<dim3(Nq / 128, M / 128), blk256, 0, stream>>>(x, Wq, q_buf, M, Nq, kH);
  gemm_bf16_kernel<<<dim3(kHD / 128, M / 128), blk256, 0, stream>>>(x, Wk, k_buf, M, kHD, kH);
  gemm_bf16_kernel<<<dim3(kHD / 128, M / 128), blk256, 0, stream>>>(x, Wv, v_buf, M, kHD, kH);
  // RMSNorm + RoPE + pack
  normrope_kernel<<<dim3(M), dim3(288), 0, stream>>>(q_buf, k_buf, v_buf, positions,
                                                     cosT, sinT, qw, kw, qh, kh, vh);
  // Flash attention
  attn_kernel<<<dim3(kB * kNH * (kT / 64)), dim3(128), 0, stream>>>(qh, kh, vh, attn_buf);
  // Output projection
  gemm_bf16_kernel<<<dim3(kH / 128, M / 128), blk256, 0, stream>>>(attn_buf, Wo, out, M, kH, Nq);
}
